// NetVLAD_37357625541244
// MI455X (gfx1250) — matrix operand order repaired and checked by reading, not compiled, not run on hardware
//
#include <hip/hip_runtime.h>
#include <hip/hip_bf16.h>

typedef __attribute__((ext_vector_type(16))) __bf16 bf16x16;
typedef __attribute__((ext_vector_type(8)))  __bf16 bf16x8;
typedef __attribute__((ext_vector_type(8)))  float  f32x8;

union FragBF { bf16x16 v; bf16x8 h[2]; };
union PackBF { unsigned int u; __bf16 b[2]; };

__device__ __forceinline__ f32x8 wmma_bf16(const bf16x16& a, const bf16x16& b, const f32x8& c) {
  // D = A(16x32) * B(32x16) + C, f32 accumulate -> v_wmma_f32_16x16x32_bf16
  return __builtin_amdgcn_wmma_f32_16x16x32_bf16(false, a, false, b, (short)0, c, false, false);
}

// CDNA5 async copy global->LDS, 16 bytes per lane, tracked by ASYNCcnt.
// VDST = VGPR with LDS byte address (flat LDS aperture keeps offset in addr[31:0]).
__device__ __forceinline__ void async_b128(unsigned lds_off, const void* gaddr) {
  asm volatile("global_load_async_to_lds_b128 %0, %1, off"
               :: "v"(lds_off), "v"(gaddr) : "memory");
}
__device__ __forceinline__ unsigned lds_off32(const void* p) {
  return (unsigned)(unsigned long long)p;
}
#define WAIT_ASYNC(n) asm volatile("s_wait_asynccnt " #n ::: "memory")

// -------------------- f32 -> bf16 conversion --------------------
__global__ void k_convert(const float* __restrict__ src, __bf16* __restrict__ dst, int n) {
  int i = blockIdx.x * blockDim.x + threadIdx.x;
  int stride = gridDim.x * blockDim.x;
  for (; i < n; i += stride) dst[i] = (__bf16)src[i];
}

// -------------------- Kernel 1: pool GEMM (D=128 x C=1024 x S-tile=112) + bias + L2norm over D ----
// A tile (Wpool bf16) double-buffered via global_load_async_to_lds_b128; x tile staged through
// VALU (needs f32->bf16 convert), overlapping the in-flight async copy.
__global__ void __launch_bounds__(256) k_pool(const float* __restrict__ x,
                                              const __bf16* __restrict__ Wp,
                                              const float* __restrict__ bpool,
                                              __bf16* __restrict__ xf) {
  constexpr int LD = 40;                            // padded bf16 row stride (80 B)
  __shared__ __align__(16) float fbuf[128 * 112];   // 57344 B, aliased for staging
  __shared__ float invb[112];
  __bf16* ldsA0 = (__bf16*)fbuf;                    // 128 x 40 bf16 (10240 B)
  __bf16* ldsA1 = (__bf16*)fbuf + 128 * LD;         // 128 x 40 bf16
  __bf16* ldsBT = (__bf16*)fbuf + 2 * 128 * LD;     // 112 x 40 bf16 (8960 B)

  const int n = blockIdx.x / 7, st = blockIdx.x % 7, s0 = st * 112;
  const int tid = threadIdx.x, lane = tid & 31, w = tid >> 5;
  const int hh = lane >> 4, r = lane & 15, m0 = w * 16;

  f32x8 acc[7] = {};

  // issue A tile for c0 = 0 (512 x 16B chunks, 2 per thread -> 2 per-wave async ops)
  {
#pragma unroll
    for (int q = 0; q < 2; ++q) {
      int id = tid + q * 256;
      int row = id >> 2, part = id & 3;
      async_b128(lds_off32(&ldsA0[row * LD + part * 8]),
                 (const void*)(Wp + row * 1024 + part * 8));
    }
  }

  int buf = 0;
  for (int it = 0; it < 32; ++it) {
    const int c0 = it * 32;
    // issue next A tile into the other buffer
    if (it + 1 < 32) {
      __bf16* nxt = buf ? ldsA0 : ldsA1;
#pragma unroll
      for (int q = 0; q < 2; ++q) {
        int id = tid + q * 256;
        int row = id >> 2, part = id & 3;
        async_b128(lds_off32(&nxt[row * LD + part * 8]),
                   (const void*)(Wp + row * 1024 + (c0 + 32) + part * 8));
      }
    }
    // stage B^T synchronously: ldsBT[s][c] = bf16(x[n][c0+c][s0+s])  (coalesced over s)
    for (int idx = tid; idx < 32 * 112; idx += 256) {
      int i = idx / 112, j = idx - i * 112;
      float v = x[(n * 1024 + c0 + i) * 784 + s0 + j];
      ldsBT[j * LD + i] = (__bf16)v;
    }
    // wait for this iteration's A tile (latest batch = 2 ops/wave still allowed in flight)
    if (it + 1 < 32) { WAIT_ASYNC(0x2); } else { WAIT_ASYNC(0x0); }
    __syncthreads();

    __bf16* curA = buf ? ldsA1 : ldsA0;
    FragBF a;
    a.h[0] = *(const bf16x8*)&curA[(m0 + r) * LD + hh * 8];
    a.h[1] = *(const bf16x8*)&curA[(m0 + r) * LD + 16 + hh * 8];
#pragma unroll
    for (int t = 0; t < 7; ++t) {
      FragBF b;
      b.h[0] = *(const bf16x8*)&ldsBT[(t * 16 + r) * LD + hh * 8];
      b.h[1] = *(const bf16x8*)&ldsBT[(t * 16 + r) * LD + 16 + hh * 8];
      acc[t] = wmma_bf16(a.v, b.v, acc[t]);
    }
    __syncthreads();
    buf ^= 1;
  }
  // epilogue: + bias, L2 norm over D per column, store bf16
#pragma unroll
  for (int t = 0; t < 7; ++t)
#pragma unroll
    for (int v = 0; v < 8; ++v) {
      int row = m0 + v + 8 * hh;
      fbuf[row * 112 + t * 16 + r] = acc[t][v] + bpool[row];
    }
  __syncthreads();
  if (tid < 112) {
    float s = 0.f;
    for (int d = 0; d < 128; ++d) { float f = fbuf[d * 112 + tid]; s += f * f; }
    invb[tid] = 1.0f / fmaxf(sqrtf(s), 1e-12f);
  }
  __syncthreads();
  for (int idx = tid; idx < 128 * 112; idx += 256) {
    int d = idx / 112, j = idx - d * 112;
    xf[(n * 128 + d) * 784 + s0 + j] = (__bf16)(fbuf[idx] * invb[j]);
  }
}

// -------------------- Kernel 2: assignment GEMM (K=64 x D=128 x S-tile) + softmax over K --------
__global__ void __launch_bounds__(128) k_assign(const __bf16* __restrict__ xf,
                                                const __bf16* __restrict__ Wc,
                                                const float* __restrict__ bconv,
                                                __bf16* __restrict__ sa) {
  constexpr int LD = 40;
  __shared__ __align__(16) float fbuf[64 * 112];   // 28672 B, aliased staging
  __shared__ float invb[112];
  __bf16* ldsA  = (__bf16*)fbuf;               // 64 x 40
  __bf16* ldsBT = (__bf16*)fbuf + 64 * LD;     // 112 x 40

  const int n = blockIdx.x / 7, st = blockIdx.x % 7, s0 = st * 112;
  const int tid = threadIdx.x, lane = tid & 31, w = tid >> 5;
  const int hh = lane >> 4, r = lane & 15, m0 = w * 16;

  f32x8 acc[7] = {};

  for (int c0 = 0; c0 < 128; c0 += 32) {
    for (int idx = tid; idx < 64 * 16; idx += 128) {
      int k = idx >> 4, cw = idx & 15;
      unsigned int u = ((const unsigned int*)(Wc + k * 128 + c0))[cw];
      *(unsigned int*)&ldsA[k * LD + 2 * cw] = u;
    }
    for (int idx = tid; idx < 32 * 56; idx += 128) {
      int i = idx / 56, jw = idx - i * 56;
      PackBF p; p.u = ((const unsigned int*)(xf + (n * 128 + c0 + i) * 784 + s0))[jw];
      ldsBT[(2 * jw) * LD + i]     = p.b[0];
      ldsBT[(2 * jw + 1) * LD + i] = p.b[1];
    }
    __syncthreads();
    FragBF a;
    a.h[0] = *(const bf16x8*)&ldsA[(m0 + r) * LD + hh * 8];
    a.h[1] = *(const bf16x8*)&ldsA[(m0 + r) * LD + 16 + hh * 8];
#pragma unroll
    for (int t = 0; t < 7; ++t) {
      FragBF b;
      b.h[0] = *(const bf16x8*)&ldsBT[(t * 16 + r) * LD + hh * 8];
      b.h[1] = *(const bf16x8*)&ldsBT[(t * 16 + r) * LD + 16 + hh * 8];
      acc[t] = wmma_bf16(a.v, b.v, acc[t]);
    }
    __syncthreads();
  }
#pragma unroll
  for (int t = 0; t < 7; ++t)
#pragma unroll
    for (int v = 0; v < 8; ++v) {
      int row = m0 + v + 8 * hh;
      fbuf[row * 112 + t * 16 + r] = acc[t][v] + bconv[row];
    }
  __syncthreads();
  if (tid < 112) {
    float mx = -1e30f;
    for (int k = 0; k < 64; ++k) mx = fmaxf(mx, fbuf[k * 112 + tid]);
    float sum = 0.f;
    for (int k = 0; k < 64; ++k) {
      float e = __expf(fbuf[k * 112 + tid] - mx);
      fbuf[k * 112 + tid] = e; sum += e;
    }
    invb[tid] = 1.0f / sum;
  }
  __syncthreads();
  for (int idx = tid; idx < 64 * 112; idx += 128) {
    int k = idx / 112, j = idx - k * 112;
    sa[(n * 64 + k) * 784 + s0 + j] = (__bf16)(fbuf[idx] * invb[j]);
  }
}

// -------------------- Kernel: asum[n,k] = sum_s sa[n,k,s] (deterministic, no atomics) ----------
__global__ void __launch_bounds__(128) k_asum(const __bf16* __restrict__ sa,
                                              float* __restrict__ asum) {
  __shared__ float red[128];
  const int row = blockIdx.x;   // n*64 + k
  float s = 0.f;
  for (int j = threadIdx.x; j < 784; j += 128) s += (float)sa[row * 784 + j];
  red[threadIdx.x] = s; __syncthreads();
  for (int o = 64; o > 0; o >>= 1) {
    if (threadIdx.x < o) red[threadIdx.x] += red[threadIdx.x + o];
    __syncthreads();
  }
  if (threadIdx.x == 0) asum[row] = red[0];
}

// -------------------- Kernel 3: vlad = sa(64x784) x xf^T(784x128) - centroids*asum, L2 over K ---
__global__ void __launch_bounds__(256) k_vlad(const __bf16* __restrict__ sa,
                                              const __bf16* __restrict__ xf,
                                              const float* __restrict__ centroids,
                                              const float* __restrict__ asum,
                                              __bf16* __restrict__ vlad) {
  constexpr int LD = 40;
  __shared__ __align__(16) float fbuf[64 * 128];   // 32768 B, aliased staging
  __shared__ float invb[128];
  __bf16* ldsA  = (__bf16*)fbuf;               // 64 x 40
  __bf16* ldsBT = (__bf16*)fbuf + 64 * LD;     // 128 x 40

  const int n = blockIdx.x;
  const int tid = threadIdx.x, lane = tid & 31, w = tid >> 5;
  const int hh = lane >> 4, r = lane & 15;
  const int m0 = (w & 3) * 16, dbase = (w >> 2) * 64;

  f32x8 acc[4] = {};

  for (int s0 = 0; s0 < 784; s0 += 32) {         // 25 chunks, last masked
    for (int idx = tid; idx < 64 * 32; idx += 256) {
      int m = idx >> 5, k = idx & 31, s = s0 + k;
      ldsA[m * LD + k] = (s < 784) ? sa[(n * 64 + m) * 784 + s] : (__bf16)0.f;
    }
    for (int idx = tid; idx < 128 * 32; idx += 256) {
      int d = idx >> 5, k = idx & 31, s = s0 + k;
      ldsBT[d * LD + k] = (s < 784) ? xf[(n * 128 + d) * 784 + s] : (__bf16)0.f;
    }
    __syncthreads();
    FragBF a;
    a.h[0] = *(const bf16x8*)&ldsA[(m0 + r) * LD + hh * 8];
    a.h[1] = *(const bf16x8*)&ldsA[(m0 + r) * LD + 16 + hh * 8];
#pragma unroll
    for (int t = 0; t < 4; ++t) {
      int col = dbase + t * 16 + r;
      FragBF b;
      b.h[0] = *(const bf16x8*)&ldsBT[col * LD + hh * 8];
      b.h[1] = *(const bf16x8*)&ldsBT[col * LD + 16 + hh * 8];
      acc[t] = wmma_bf16(a.v, b.v, acc[t]);
    }
    __syncthreads();
  }
#pragma unroll
  for (int t = 0; t < 4; ++t)
#pragma unroll
    for (int v = 0; v < 8; ++v) {
      int row = m0 + v + 8 * hh, col = dbase + t * 16 + r;
      fbuf[row * 128 + col] = acc[t][v];
    }
  __syncthreads();
  if (tid < 128) {
    float ss = 0.f;
    for (int k = 0; k < 64; ++k) {
      float val = fbuf[k * 128 + tid] - centroids[k * 128 + tid] * asum[n * 64 + k];
      fbuf[k * 128 + tid] = val; ss += val * val;
    }
    invb[tid] = 1.0f / fmaxf(sqrtf(ss), 1e-12f);
  }
  __syncthreads();
  for (int idx = tid; idx < 8192; idx += 256)
    vlad[n * 8192 + idx] = (__bf16)(fbuf[idx] * invb[idx & 127]);
}

// -------------------- Kernel 4: projection 32 x 8192 x 1024 + bias -> pre (f32) -----------------
// Fully async double-buffered staging (A = vlad tile, B^T = Wproj tile), 256 K-chunks.
__global__ void __launch_bounds__(256) k_proj(const __bf16* __restrict__ vlad,
                                              const __bf16* __restrict__ Wpj,
                                              const float* __restrict__ bproj,
                                              float* __restrict__ pre) {
  constexpr int LD = 40;
  __shared__ __align__(16) __bf16 lds[2 * 32 * LD + 2 * 128 * LD];   // 25600 B
  __bf16* a0  = lds;                       // 32 x 40
  __bf16* a1  = lds + 32 * LD;
  __bf16* bt0 = lds + 2 * 32 * LD;         // 128 x 40
  __bf16* bt1 = lds + 2 * 32 * LD + 128 * LD;

  const int obase = blockIdx.x * 128;
  const int tid = threadIdx.x, lane = tid & 31, w = tid >> 5;
  const int hh = lane >> 4, r = lane & 15;

  f32x8 acc0 = {}, acc1 = {};

  // issue batch 0:  A: 128 chunks (waves 0-3, 1 op/wave), B^T: 512 chunks (2 ops/wave)
  {
    if (tid < 128) {
      int row = tid >> 2, part = tid & 3;
      async_b128(lds_off32(&a0[row * LD + part * 8]),
                 (const void*)(vlad + row * 8192 + part * 8));
    }
#pragma unroll
    for (int q = 0; q < 2; ++q) {
      int id = tid + q * 256;
      int row = id >> 2, part = id & 3;
      async_b128(lds_off32(&bt0[row * LD + part * 8]),
                 (const void*)(Wpj + (size_t)(obase + row) * 8192 + part * 8));
    }
  }

  int buf = 0;
  for (int it = 0; it < 256; ++it) {
    const int j0 = it * 32;
    if (it + 1 < 256) {
      __bf16* na  = buf ? a0 : a1;
      __bf16* nbt = buf ? bt0 : bt1;
      if (tid < 128) {
        int row = tid >> 2, part = tid & 3;
        async_b128(lds_off32(&na[row * LD + part * 8]),
                   (const void*)(vlad + row * 8192 + (j0 + 32) + part * 8));
      }
#pragma unroll
      for (int q = 0; q < 2; ++q) {
        int id = tid + q * 256;
        int row = id >> 2, part = id & 3;
        async_b128(lds_off32(&nbt[row * LD + part * 8]),
                   (const void*)(Wpj + (size_t)(obase + row) * 8192 + (j0 + 32) + part * 8));
      }
      // latest batch: waves 0-3 issued 3 ops, waves 4-7 issued 2 -> wait for previous batch
      if (w < 4) { WAIT_ASYNC(0x3); } else { WAIT_ASYNC(0x2); }
    } else {
      WAIT_ASYNC(0x0);
    }
    __syncthreads();

    __bf16* A  = buf ? a1 : a0;
    __bf16* BT = buf ? bt1 : bt0;
    FragBF fa0, fa1, b;
    fa0.h[0] = *(const bf16x8*)&A[r * LD + hh * 8];
    fa0.h[1] = *(const bf16x8*)&A[r * LD + 16 + hh * 8];
    fa1.h[0] = *(const bf16x8*)&A[(16 + r) * LD + hh * 8];
    fa1.h[1] = *(const bf16x8*)&A[(16 + r) * LD + 16 + hh * 8];
    int ol = w * 16 + r;
    b.h[0] = *(const bf16x8*)&BT[ol * LD + hh * 8];
    b.h[1] = *(const bf16x8*)&BT[ol * LD + 16 + hh * 8];
    acc0 = wmma_bf16(fa0.v, b.v, acc0);
    acc1 = wmma_bf16(fa1.v, b.v, acc1);
    __syncthreads();
    buf ^= 1;
  }
  int o = obase + w * 16 + r;
  float bias = bproj[o];
#pragma unroll
  for (int v = 0; v < 8; ++v) {
    pre[(v + 8 * hh) * 1024 + o]      = acc0[v] + bias;
    pre[(16 + v + 8 * hh) * 1024 + o] = acc1[v] + bias;
  }
}

// -------------------- Kernel 5: final per-row L2 norm -> d_out --------------------
__global__ void __launch_bounds__(256) k_final(const float* __restrict__ pre,
                                               float* __restrict__ out) {
  __shared__ float red[256];
  const int n = blockIdx.x;
  float s = 0.f;
  for (int o = threadIdx.x; o < 1024; o += 256) { float f = pre[n * 1024 + o]; s += f * f; }
  red[threadIdx.x] = s; __syncthreads();
  for (int off = 128; off > 0; off >>= 1) {
    if (threadIdx.x < off) red[threadIdx.x] += red[threadIdx.x + off];
    __syncthreads();
  }
  float inv = 1.0f / fmaxf(sqrtf(red[0]), 1e-12f);
  for (int o = threadIdx.x; o < 1024; o += 256) out[n * 1024 + o] = pre[n * 1024 + o] * inv;
}

// -------------------- Workspace layout (bytes, 256-aligned) --------------------
static constexpr size_t OFF_WPOOL = 0;                       // 131072 bf16  = 262144 B
static constexpr size_t OFF_WCONV = 262144;                  // 8192 bf16    = 16384 B
static constexpr size_t OFF_WPROJ = 278528;                  // 8388608 bf16 = 16777216 B
static constexpr size_t OFF_XF    = 17055744;                // 32*128*784 bf16 = 6422528 B
static constexpr size_t OFF_SA    = 23478272;                // 32*64*784 bf16  = 3211264 B
static constexpr size_t OFF_ASUM  = 26689536;                // 2048 f32     = 8192 B
static constexpr size_t OFF_VLAD  = 26697728;                // 32*8192 bf16 = 524288 B
static constexpr size_t OFF_PRE   = 27222016;                // 32*1024 f32  = 131072 B
// total = 27353088 B (~26.1 MiB)

extern "C" void kernel_launch(void* const* d_in, const int* in_sizes, int n_in,
                              void* d_out, int out_size, void* d_ws, size_t ws_size,
                              hipStream_t stream) {
  (void)in_sizes; (void)n_in; (void)out_size; (void)ws_size;
  const float* x         = (const float*)d_in[0];
  const float* Wpool     = (const float*)d_in[1];
  const float* bpool     = (const float*)d_in[2];
  const float* Wconv     = (const float*)d_in[3];
  const float* bconv     = (const float*)d_in[4];
  const float* centroids = (const float*)d_in[5];
  const float* Wproj     = (const float*)d_in[6];
  const float* bproj     = (const float*)d_in[7];
  float* out = (float*)d_out;
  char* ws = (char*)d_ws;

  __bf16* wpool_bf = (__bf16*)(ws + OFF_WPOOL);
  __bf16* wconv_bf = (__bf16*)(ws + OFF_WCONV);
  __bf16* wproj_bf = (__bf16*)(ws + OFF_WPROJ);
  __bf16* xf_bf    = (__bf16*)(ws + OFF_XF);
  __bf16* sa_bf    = (__bf16*)(ws + OFF_SA);
  float*  asum     = (float*)(ws + OFF_ASUM);
  __bf16* vlad_bf  = (__bf16*)(ws + OFF_VLAD);
  float*  pre      = (float*)(ws + OFF_PRE);

  k_convert<<<512, 256, 0, stream>>>(Wpool, wpool_bf, 131072);
  k_convert<<<32, 256, 0, stream>>>(Wconv, wconv_bf, 8192);
  k_convert<<<4096, 256, 0, stream>>>(Wproj, wproj_bf, 8388608);

  k_pool<<<32 * 7, 256, 0, stream>>>(x, wpool_bf, bpool, xf_bf);
  k_assign<<<32 * 7, 128, 0, stream>>>(xf_bf, wconv_bf, bconv, sa_bf);
  k_asum<<<32 * 64, 128, 0, stream>>>(sa_bf, asum);
  k_vlad<<<32, 256, 0, stream>>>(sa_bf, xf_bf, centroids, asum, vlad_bf);
  k_proj<<<8, 256, 0, stream>>>(vlad_bf, wproj_bf, bproj, pre);
  k_final<<<32, 256, 0, stream>>>(pre, out);
}
